// CGC_layer_5291399709045
// MI455X (gfx1250) — matrix operand order repaired and checked by reading, not compiled, not run on hardware
//
#include <hip/hip_runtime.h>

// ---------------------------------------------------------------------------
// CDNA5 (gfx1250) expert-MLP + gating.
// bf16 WMMA (v_wmma_f32_16x16x32_bf16) fed by the Tensor Data Mover
// (tensor_load_to_lds + s_wait_tensorcnt), double-buffered LDS.
// Shapes: B=8192 D=2048 H1=1024 H2=512 G=512 T=2 SPEC=2 E=6 EC=4
// ---------------------------------------------------------------------------

#define B_  8192
#define D_  2048
#define H1_ 1024
#define H2_ 512
#define G_  512
#define T_  2
#define E_  6
#define EC_ 4

typedef __attribute__((ext_vector_type(16))) __bf16    v16bf;
typedef __attribute__((ext_vector_type(8)))  float     v8f;
typedef __attribute__((ext_vector_type(4)))  unsigned  uv4;
typedef __attribute__((ext_vector_type(8)))  unsigned  uv8;

union FragBF {
    v16bf v;
    uint4 q[2];
};

__device__ __forceinline__ unsigned short f2bf(float f) {
    unsigned u = __float_as_uint(f);
    u += 0x7FFFu + ((u >> 16) & 1u);     // round-to-nearest-even
    return (unsigned short)(u >> 16);
}
__device__ __forceinline__ unsigned pack2(float lo, float hi) {
    return (unsigned)f2bf(lo) | ((unsigned)f2bf(hi) << 16);
}

// ---------------------------------------------------------------------------
// TDM: DMA one [rows=128 x cols=32] bf16 tile (row stride K elements) from
// global into LDS at lds_addr, hardware-padding each 64B row with 16B so the
// LDS row stride is 80B (matches fragment loader).  2-D tensor -> 2 descriptor
// groups, issued by one wave; tracked by TENSORcnt.
// ---------------------------------------------------------------------------
__device__ __forceinline__ void tdm_tile_load(const void* gptr, unsigned lds_addr,
                                              int K) {
    const unsigned long long ga = (unsigned long long)(uintptr_t)gptr;
    uv4 g0;
    g0[0] = 1u;                                   // count=1, user descriptor
    g0[1] = lds_addr;                             // LDS byte address
    g0[2] = (unsigned)ga;                         // global_addr[31:0]
    g0[3] = ((unsigned)(ga >> 32) & 0x01FFFFFFu)  // global_addr[56:32]
            | 0x80000000u;                        // type=2 ("image")
    uv8 g1;
    // data_size=1(2B) | pad_enable | pad_interval=3(16 DW=64B) | pad_amount=3(4 DW=16B)
    g1[0] = 0x06D10000u;
    g1[1] = 0xFFFF0000u;                          // tensor_dim0 = 0x7FFFFFFF (lo16<<16)
    g1[2] = 0xFFFF7FFFu;                          // tensor_dim0 hi16 | tensor_dim1 lo16
    g1[3] = 0x7FFFu | (32u << 16);                // tensor_dim1 hi16 | tile_dim0=32
    g1[4] = 128u;                                 // tile_dim1=128, tile_dim2=0
    g1[5] = (unsigned)K;                          // tensor_dim0_stride[31:0] (elements)
    g1[6] = 0u;                                   // stride0 hi16 | stride1 lo16
    g1[7] = 0u;                                   // stride1 hi
    asm volatile("tensor_load_to_lds %0, %1" :: "s"(g0), "s"(g1) : "memory");
}

// ---------------------------------------------------------------------------
// C[M x N] = relu(A[M x K] @ Wt[N x K]^T + bias[N])
// A, Wt: bf16 row-major in global.  OBF: store C as bf16, else f32.
// blockIdx.x -> M/128, blockIdx.y -> N/128, blockIdx.z -> expert e.
// ---------------------------------------------------------------------------
#define LDR 20            // LDS row stride in uints (16 data + 4 pad) = 80 B
#define TILEU (128 * LDR) // uints per tile buffer

template <bool OBF>
__global__ __launch_bounds__(256)
void gemm_bf16_tdm(const unsigned short* __restrict__ A, size_t aEStride,
                   const unsigned short* __restrict__ Wt, size_t wEStride,
                   const float* __restrict__ bias, size_t bEStride,
                   void* __restrict__ Cv, size_t cEStride,
                   int K, int N)
{
    const int e = blockIdx.z;
    A    += (size_t)e * aEStride;
    Wt   += (size_t)e * wEStride;
    bias += (size_t)e * bEStride;

    const int tid  = threadIdx.x;
    const int lane = tid & 31;
    const int wave = tid >> 5;
    const int wm   = wave & 1;   // 2 waves along M (64 rows each)
    const int wn   = wave >> 1;  // 4 waves along N (32 cols each)
    const int lr   = lane & 15;
    const int h    = lane >> 4;  // lane half

    const int blockM = blockIdx.x * 128;
    const int blockN = blockIdx.y * 128;

    __shared__ __align__(16) unsigned As[2][TILEU];
    __shared__ __align__(16) unsigned Bs[2][TILEU];
    const unsigned ldsA0 = (unsigned)(uintptr_t)(void*)&As[0][0];
    const unsigned ldsA1 = (unsigned)(uintptr_t)(void*)&As[1][0];
    const unsigned ldsB0 = (unsigned)(uintptr_t)(void*)&Bs[0][0];
    const unsigned ldsB1 = (unsigned)(uintptr_t)(void*)&Bs[1][0];

    const unsigned short* Abase = A  + (size_t)blockM * K;   // tile row 0
    const unsigned short* Bbase = Wt + (size_t)blockN * K;

    v8f acc[4][2] = {};

    const int ktiles = K >> 5;

    // prime buffer 0
    if (wave == 0) {
        tdm_tile_load(Abase, ldsA0, K);
        tdm_tile_load(Bbase, ldsB0, K);
    }

    int cur = 0;
    for (int kt = 0; kt < ktiles; ++kt) {
        if (wave == 0)
            __builtin_amdgcn_s_wait_tensorcnt(0);   // tiles for kt landed
        __syncthreads();                            // publish LDS to all waves

        // overlap: DMA tile kt+1 into the other buffer while we compute kt
        if (wave == 0 && (kt + 1) < ktiles) {
            const size_t off = (size_t)(kt + 1) * 32;
            tdm_tile_load(Abase + off, cur ? ldsA0 : ldsA1, K);
            tdm_tile_load(Bbase + off, cur ? ldsB0 : ldsB1, K);
        }

        const unsigned* AsB = &As[cur][0];
        const unsigned* BsB = &Bs[cur][0];

        // fragments per ISA VGPR layout (16-bit A 16x32 / B 32x16)
        FragBF fa[4], fb[2];
        #pragma unroll
        for (int i = 0; i < 4; ++i) {
            const unsigned* row = AsB + (wm * 64 + i * 16 + lr) * LDR;
            fa[i].q[0] = *reinterpret_cast<const uint4*>(row + h * 4);
            fa[i].q[1] = *reinterpret_cast<const uint4*>(row + 8 + h * 4);
        }
        #pragma unroll
        for (int j = 0; j < 2; ++j) {
            const unsigned* row = BsB + (wn * 32 + j * 16 + lr) * LDR;
            fb[j].q[0] = *reinterpret_cast<const uint4*>(row + h * 4);
            fb[j].q[1] = *reinterpret_cast<const uint4*>(row + 8 + h * 4);
        }
        #pragma unroll
        for (int i = 0; i < 4; ++i)
            #pragma unroll
            for (int j = 0; j < 2; ++j)
                acc[i][j] = __builtin_amdgcn_wmma_f32_16x16x32_bf16(
                    false, fa[i].v, false, fb[j].v, (short)0, acc[i][j],
                    false, false);

        __syncthreads();    // compute(kt) done before buffer is overwritten
        cur ^= 1;
    }

    // ---- epilogue: bias + relu (C layout: VGPR v -> M = v + 8h, N = lr) ----
    #pragma unroll
    for (int i = 0; i < 4; ++i) {
        #pragma unroll
        for (int j = 0; j < 2; ++j) {
            const int n = blockN + wn * 32 + j * 16 + lr;
            const float bv = bias[n];
            #pragma unroll
            for (int v = 0; v < 8; ++v) {
                const int m = blockM + wm * 64 + i * 16 + h * 8 + v;
                float val = acc[i][j][v] + bv;
                val = val > 0.f ? val : 0.f;
                if (OBF) {
                    unsigned short* C = (unsigned short*)Cv + (size_t)e * cEStride;
                    C[(size_t)m * N + n] = f2bf(val);
                } else {
                    float* C = (float*)Cv + (size_t)e * cEStride;
                    C[(size_t)m * N + n] = val;
                }
            }
        }
    }
}

// ---------------------------------------------------------------------------
// prep kernels: one-time f32 -> bf16 conversion (and weight transpose)
// ---------------------------------------------------------------------------

// out[i] = bf16(in[i]), 8 elements per thread, fully vectorized
__global__ __launch_bounds__(256)
void cvt_bf16_kernel(const float* __restrict__ in, unsigned short* __restrict__ outp)
{
    const size_t i = ((size_t)blockIdx.x * 256 + threadIdx.x) * 8;
    const float4 a = *reinterpret_cast<const float4*>(in + i);
    const float4 b = *reinterpret_cast<const float4*>(in + i + 4);
    uint4 o;
    o.x = pack2(a.x, a.y); o.y = pack2(a.z, a.w);
    o.z = pack2(b.x, b.y); o.w = pack2(b.z, b.w);
    *reinterpret_cast<uint4*>(outp + i) = o;
}

// Wt[e][n][k] = bf16(W[e][k][n]); thread handles 8 k for one n
__global__ __launch_bounds__(256)
void transpose_cvt_kernel(const float* __restrict__ W, unsigned short* __restrict__ Wt,
                          int K, int N)
{
    const int e = blockIdx.z;
    W  += (size_t)e * K * N;
    Wt += (size_t)e * N * K;
    const int idx = blockIdx.x * 256 + threadIdx.x;
    const int n  = idx % N;
    const int kb = (idx / N) * 8;
    uint4 o;
    o.x = pack2(W[(size_t)(kb + 0) * N + n], W[(size_t)(kb + 1) * N + n]);
    o.y = pack2(W[(size_t)(kb + 2) * N + n], W[(size_t)(kb + 3) * N + n]);
    o.z = pack2(W[(size_t)(kb + 4) * N + n], W[(size_t)(kb + 5) * N + n]);
    o.w = pack2(W[(size_t)(kb + 6) * N + n], W[(size_t)(kb + 7) * N + n]);
    *reinterpret_cast<uint4*>(Wt + (size_t)n * K + kb) = o;
}

// ---------------------------------------------------------------------------
// gate[b][t][:] = softmax(g[t][b][:] @ Wgs[t])  -- one wave per (t,b) row
// ---------------------------------------------------------------------------
__global__ __launch_bounds__(256)
void gate_kernel(const float* __restrict__ g, const float* __restrict__ Wgs,
                 float* __restrict__ gate)
{
    const int lane = threadIdx.x & 31;
    const int wid  = blockIdx.x * 8 + (threadIdx.x >> 5);  // over [T][B]
    const int t = wid >> 13;       // B_ = 2^13
    const int b = wid & (B_ - 1);
    const float* grow = g + ((size_t)t * B_ + b) * G_;
    const float* wt   = Wgs + (size_t)t * G_ * EC_;

    float a0 = 0.f, a1 = 0.f, a2 = 0.f, a3 = 0.f;
    for (int gg = lane; gg < G_; gg += 32) {
        const float gv = grow[gg];
        const float4 w = *reinterpret_cast<const float4*>(wt + gg * 4);
        a0 += gv * w.x; a1 += gv * w.y; a2 += gv * w.z; a3 += gv * w.w;
    }
    #pragma unroll
    for (int off = 16; off > 0; off >>= 1) {
        a0 += __shfl_xor(a0, off);
        a1 += __shfl_xor(a1, off);
        a2 += __shfl_xor(a2, off);
        a3 += __shfl_xor(a3, off);
    }
    if (lane == 0) {
        const float m  = fmaxf(fmaxf(a0, a1), fmaxf(a2, a3));
        const float e0 = __expf(a0 - m), e1 = __expf(a1 - m);
        const float e2 = __expf(a2 - m), e3 = __expf(a3 - m);
        const float inv = 1.f / (e0 + e1 + e2 + e3);
        float4 r; r.x = e0 * inv; r.y = e1 * inv; r.z = e2 * inv; r.w = e3 * inv;
        *reinterpret_cast<float4*>(gate + ((size_t)b * T_ + t) * EC_) = r;
    }
}

// ---------------------------------------------------------------------------
// out[t][b][o] = sum_j gate[b][t][j] * h2[idx[t][j]][b][o], idx[t]={2t,2t+1,4,5}
// ---------------------------------------------------------------------------
__global__ __launch_bounds__(256)
void combine_kernel(const float* __restrict__ h2, const float* __restrict__ gate,
                    float* __restrict__ out)
{
    const size_t i = (size_t)blockIdx.x * 256 + threadIdx.x;  // [T][B][H2]
    const int o = (int)(i & (H2_ - 1));
    const int b = (int)((i >> 9) & (B_ - 1));   // H2_ = 2^9
    const int t = (int)(i >> 22);               // B_*H2_ = 2^22
    const float* gt = gate + ((size_t)b * T_ + t) * EC_;
    float s = 0.f;
    #pragma unroll
    for (int j = 0; j < 4; ++j) {
        const int e = (j < 2) ? (t * 2 + j) : (j + 2);
        s += gt[j] * h2[((size_t)e * B_ + b) * H2_ + o];
    }
    out[i] = s;
}

// ---------------------------------------------------------------------------
extern "C" void kernel_launch(void* const* d_in, const int* in_sizes, int n_in,
                              void* d_out, int out_size, void* d_ws, size_t ws_size,
                              hipStream_t stream) {
    const float* x   = (const float*)d_in[0];
    const float* We1 = (const float*)d_in[1];
    const float* be1 = (const float*)d_in[2];
    const float* We2 = (const float*)d_in[3];
    const float* be2 = (const float*)d_in[4];
    const float* Wg1 = (const float*)d_in[5];
    const float* bg1 = (const float*)d_in[6];
    const float* Wgs = (const float*)d_in[7];
    float* out = (float*)d_out;

    // workspace carve-up (bf16 buffers as ushort); all written before read
    char* p = (char*)d_ws;
    unsigned short* xb   = (unsigned short*)p; p += (size_t)B_ * D_ * 2;        // [B][D]
    unsigned short* We1t = (unsigned short*)p; p += (size_t)E_ * H1_ * D_ * 2;  // [E][H1][D]
    unsigned short* We2t = (unsigned short*)p; p += (size_t)E_ * H2_ * H1_ * 2; // [E][H2][H1]
    unsigned short* Wg1t = (unsigned short*)p; p += (size_t)T_ * G_ * D_ * 2;   // [T][G][D]
    unsigned short* h1   = (unsigned short*)p; p += (size_t)E_ * B_ * H1_ * 2;  // [E][B][H1]
    float* h2   = (float*)p; p += (size_t)E_ * B_ * H2_ * 4;                    // [E][B][H2]
    float* g    = (float*)p; p += (size_t)T_ * B_ * G_ * 4;                     // [T][B][G]
    float* gate = (float*)p;                                                    // [B][T][EC]

    dim3 blk(256);

    // ---- prep: f32 -> bf16 (+ weight transpose to [N][K]) ----
    cvt_bf16_kernel<<<dim3(((size_t)B_ * D_) / (8 * 256)), blk, 0, stream>>>(x, xb);
    transpose_cvt_kernel<<<dim3(((size_t)H1_ * (D_ / 8)) / 256, 1, E_), blk, 0, stream>>>(
        We1, We1t, D_, H1_);
    transpose_cvt_kernel<<<dim3(((size_t)H2_ * (H1_ / 8)) / 256, 1, E_), blk, 0, stream>>>(
        We2, We2t, H1_, H2_);
    transpose_cvt_kernel<<<dim3(((size_t)G_ * (D_ / 8)) / 256, 1, T_), blk, 0, stream>>>(
        Wg1, Wg1t, D_, G_);

    // ---- h1 = relu(x @ We1[e] + be1[e])  (bf16 out) ----
    gemm_bf16_tdm<true><<<dim3(B_ / 128, H1_ / 128, E_), blk, 0, stream>>>(
        xb, (size_t)0,
        We1t, (size_t)H1_ * D_,
        be1, (size_t)H1_,
        h1, (size_t)B_ * H1_,
        D_, H1_);

    // ---- g = relu(x @ Wg1[t] + bg1[t])  (f32 out) ----
    gemm_bf16_tdm<false><<<dim3(B_ / 128, G_ / 128, T_), blk, 0, stream>>>(
        xb, (size_t)0,
        Wg1t, (size_t)G_ * D_,
        bg1, (size_t)G_,
        g, (size_t)B_ * G_,
        D_, G_);

    // ---- h2 = relu(h1[e] @ We2[e] + be2[e])  (f32 out) ----
    gemm_bf16_tdm<false><<<dim3(B_ / 128, H2_ / 128, E_), blk, 0, stream>>>(
        h1, (size_t)B_ * H1_,
        We2t, (size_t)H2_ * H1_,
        be2, (size_t)H2_,
        h2, (size_t)B_ * H2_,
        H1_, H2_);

    // ---- gate = softmax(g @ Wgs) ----
    gate_kernel<<<dim3((T_ * B_) / 8), blk, 0, stream>>>(g, Wgs, gate);

    // ---- out = weighted combine of selected experts ----
    combine_kernel<<<dim3(((size_t)T_ * B_ * H2_) / 256), blk, 0, stream>>>(
        h2, gate, out);
}
